// MultiHeadDistanceLayer_2370821947428
// MI455X (gfx1250) — matrix operand order repaired and checked by reading, not compiled, not run on hardware
//
#include <hip/hip_runtime.h>
#include <hip/hip_bf16.h>
#include <math.h>

// Problem constants (from reference)
#define B_  4
#define L_  1024
#define D_  256
#define H_  8
#define HD_ 64

typedef __attribute__((ext_vector_type(16))) _Float16 v16h;
typedef __attribute__((ext_vector_type(8)))  _Float16 v8h;
typedef __attribute__((ext_vector_type(8)))  float    v8f;

// ---------------------------------------------------------------------------
// Kernel 0: prep — f16 convert (x+pe) -> Ah [4096 x 256],
//                  build WT = [Wq|Wk]^T in f16 [1024 x 256] (K-contiguous rows),
//                  zero the diagonal accumulator S [H*B*L].
// Grid: 4096 x 256 threads = 1,048,576 = B*L*D elements.
// ---------------------------------------------------------------------------
__global__ __launch_bounds__(256) void prep_kernel(
    const float* __restrict__ x, const float* __restrict__ pe,
    const float* __restrict__ Wq, const float* __restrict__ Wk,
    _Float16* __restrict__ Ah, _Float16* __restrict__ Wh,
    float* __restrict__ Sg)
{
    const int i = blockIdx.x * 256 + threadIdx.x;              // < B*L*D
    Ah[i] = (_Float16)(x[i] + pe[i & (L_ * D_ - 1)]);          // pe broadcast over batch
    if (i < 1024 * 256) {                                      // WT: row n (output col), col kk (reduction)
        const int n = i >> 8, kk = i & 255;
        Wh[i] = (_Float16)(n < 512 ? Wq[kk * 512 + n] : Wk[kk * 512 + (n - 512)]);
    }
    if (i < H_ * B_ * L_) Sg[i] = 0.0f;
}

// ---------------------------------------------------------------------------
// Kernel 1: QK projection GEMM via WMMA.
// C[4096 x 1024] (f16) = Ah[4096 x 256] * [Wq|Wk][256 x 1024] + bias.
// One wave computes a 16(M) x 64(N) tile: 4 f32 accumulators, 8 k-steps of 32.
// Lane layouts follow CDNA5 16-bit WMMA A/B/C VGPR tables (wave32).
// ---------------------------------------------------------------------------
__global__ __launch_bounds__(128) void gemm_qk(
    const _Float16* __restrict__ A, const _Float16* __restrict__ W,
    const float* __restrict__ bq, const float* __restrict__ bk,
    _Float16* __restrict__ C)
{
    const int wid  = blockIdx.x * 4 + (threadIdx.x >> 5);      // 4096 waves total
    const int lane = threadIdx.x & 31;
    const int li   = lane & 15;                                 // N / M index within tile
    const int hi   = lane >> 4;                                 // half select
    const int mt   = wid >> 4;                                  // 0..255 (M tiles)
    const int ng   = wid & 15;                                  // 0..15  (64-wide N groups)

    const _Float16* arow = A + (size_t)(mt * 16 + li) * 256;

    v8f acc[4] = {};
    for (int kk = 0; kk < 8; ++kk) {
        const int kb = kk * 32;
        // A operand: lane<16 holds K {0..7, 16..23}, lane>=16 holds K {8..15, 24..31}
        v16h a;
        {
            v8h lo = *(const v8h*)(arow + kb + hi * 8);
            v8h hh = *(const v8h*)(arow + kb + 16 + hi * 8);
            for (int q = 0; q < 8; ++q) { a[q] = lo[q]; a[8 + q] = hh[q]; }
        }
        if (kk < 7) __builtin_prefetch(arow + kb + 32, 0, 0);   // global_prefetch_b8
        for (int t = 0; t < 4; ++t) {
            const int col = ng * 64 + t * 16 + li;
            // B operand: column `col` of W == row `col` of WT (K-contiguous).
            // lane<16 -> K {0..7, 16..23}, lane>=16 -> K {8..15, 24..31}.
            v16h bv;
            {
                v8h bl = *(const v8h*)(W + (size_t)col * 256 + kb + hi * 8);
                v8h bh = *(const v8h*)(W + (size_t)col * 256 + kb + 16 + hi * 8);
                for (int q = 0; q < 8; ++q) { bv[q] = bl[q]; bv[8 + q] = bh[q]; }
            }
            acc[t] = __builtin_amdgcn_wmma_f32_16x16x32_f16(
                false, a, false, bv, (short)0, acc[t], false, false);
        }
    }
    // C layout: VGPR v -> M = v + 8*hi, N = li.  Add bias, store f16.
    for (int t = 0; t < 4; ++t) {
        const int col  = ng * 64 + t * 16 + li;
        const float bias = (col < 512) ? bq[col] : bk[col - 512];
        for (int v = 0; v < 8; ++v) {
            const int row = mt * 16 + v + 8 * hi;
            C[(size_t)row * 1024 + col] = (_Float16)(acc[t][v] + bias);
        }
    }
}

// ---------------------------------------------------------------------------
// Kernel 2: v'[h,b,r] = sigmoid( x[b, L-1-r, :] . Wv[:, h] )
// ---------------------------------------------------------------------------
__global__ __launch_bounds__(256) void v_kernel(
    const float* __restrict__ x, const float* __restrict__ Wv,
    float* __restrict__ vbuf)
{
    const int t  = blockIdx.x * 256 + threadIdx.x;              // < H*B*L
    const int hb = t >> 10, r = t & 1023;
    const int h  = hb >> 2, b = hb & 3;
    const int l  = (L_ - 1) - r;
    const float* xr = x + (size_t)(b * L_ + l) * D_;
    float acc = 0.0f;
    for (int d = 0; d < D_; ++d) acc = fmaf(xr[d], Wv[d * H_ + h], acc);
    vbuf[t] = 1.0f / (1.0f + __expf(-acc));
}

// ---------------------------------------------------------------------------
// Kernel 3: fused attention.
//   Per wave: 16 query rows of one (h,b).
//   pass1: scores = Q K^T / 8 via WMMA, stash f16 in LDS, track row max
//   pass2: exp + row sums (softmax denominator)
//   pass3: accumulate S[j] = sum_{k-q=j, k>=q} softmax(q,k) * v'[k]
//   Dynamic LDS: 8 waves * 16*1024 f16 exp-buffers (256 KB) + shared S (4 KB)
//   — exploits the 320 KB/WGP CDNA5 LDS.
// Grid: (H*B, L/128), 256 threads (8 waves).
// ---------------------------------------------------------------------------
__global__ __launch_bounds__(256) void attn_kernel(
    const _Float16* __restrict__ C,     // [4096 x 1024] f16, cols 0..511 = Q, 512..1023 = K
    const float* __restrict__ vbuf,     // [H*B, L] v'
    float* __restrict__ Sg)             // [H*B, L] diagonal sums (pre-zeroed)
{
    extern __shared__ char smem_raw[];
    _Float16* ebuf = (_Float16*)smem_raw;                       // 8 * 16 * 1024 f16
    float*    Sloc = (float*)(smem_raw + 8 * 16 * 1024 * 2);    // 1024 f32

    const int hb = blockIdx.x;            // h*B + b
    const int qc = blockIdx.y;            // 128-row chunk
    const int h  = hb >> 2, b = hb & 3;
    const int tid  = threadIdx.x;
    const int w    = tid >> 5;
    const int lane = tid & 31;
    const int li   = lane & 15;
    const int hi   = lane >> 4;

    for (int i = tid; i < 1024; i += 256) Sloc[i] = 0.0f;
    __syncthreads();

    const int qbase = qc * 128 + w * 16;

    // Q A-operands for both k-steps of the HD=64 reduction.
    const _Float16* qrow = C + (size_t)(b * L_ + qbase + li) * 1024 + h * HD_;
    v16h a0, a1;
    {
        v8h l0 = *(const v8h*)(qrow +      hi * 8);
        v8h h0 = *(const v8h*)(qrow + 16 + hi * 8);
        v8h l1 = *(const v8h*)(qrow + 32 + hi * 8);
        v8h h1 = *(const v8h*)(qrow + 48 + hi * 8);
        for (int q = 0; q < 8; ++q) {
            a0[q] = l0[q]; a0[8 + q] = h0[q];
            a1[q] = l1[q]; a1[8 + q] = h1[q];
        }
    }

    _Float16* mye = ebuf + (size_t)w * 16 * 1024;

    // ---- pass 1: WMMA scores -> LDS, running row max ----
    float m8[8];
    for (int v = 0; v < 8; ++v) m8[v] = -1.0e30f;

    for (int kt = 0; kt < 64; ++kt) {
        const _Float16* krow =
            C + (size_t)(b * L_ + kt * 16 + li) * 1024 + 512 + h * HD_;
        v16h b0, b1;
        {
            v8h bl0 = *(const v8h*)(krow +      hi * 8);
            v8h bh0 = *(const v8h*)(krow + 16 + hi * 8);
            v8h bl1 = *(const v8h*)(krow + 32 + hi * 8);
            v8h bh1 = *(const v8h*)(krow + 48 + hi * 8);
            for (int q = 0; q < 8; ++q) {
                b0[q] = bl0[q]; b0[8 + q] = bh0[q];
                b1[q] = bl1[q]; b1[8 + q] = bh1[q];
            }
        }
        v8f c = {};
        c = __builtin_amdgcn_wmma_f32_16x16x32_f16(false, a0, false, b0, (short)0, c, false, false);
        c = __builtin_amdgcn_wmma_f32_16x16x32_f16(false, a1, false, b1, (short)0, c, false, false);
        for (int v = 0; v < 8; ++v) {
            const float s = c[v] * 0.125f;          // 1/sqrt(HD)
            m8[v] = fmaxf(m8[v], s);
            mye[(v + 8 * hi) * 1024 + kt * 16 + li] = (_Float16)s;
        }
    }
    // Row max: reduce across the 16 lanes (N dimension) sharing each row group.
    for (int off = 1; off < 16; off <<= 1)
        for (int v = 0; v < 8; ++v)
            m8[v] = fmaxf(m8[v], __shfl_xor(m8[v], off, 32));

    // ---- pass 2: exp + row sums ----
    float s8[8];
    for (int v = 0; v < 8; ++v) s8[v] = 0.0f;
    for (int kt = 0; kt < 64; ++kt) {
        for (int v = 0; v < 8; ++v) {
            const int idx = (v + 8 * hi) * 1024 + kt * 16 + li;
            const float e = __expf((float)mye[idx] - m8[v]);
            mye[idx] = (_Float16)e;
            s8[v] += e;
        }
    }
    for (int off = 1; off < 16; off <<= 1)
        for (int v = 0; v < 8; ++v)
            s8[v] += __shfl_xor(s8[v], off, 32);
    float rinv[8];
    for (int v = 0; v < 8; ++v) rinv[v] = 1.0f / s8[v];

    // ---- pass 3: diagonal accumulation S[j] += att[q,k]*v'[k], j = k - q >= 0 ----
    const float* vp = vbuf + (size_t)hb * L_;
    for (int kt = 0; kt < 64; ++kt) {
        const int k  = kt * 16 + li;
        const float vk = vp[k];
        for (int v = 0; v < 8; ++v) {
            const int M = v + 8 * hi;
            const int j = k - (qbase + M);
            if (j >= 0) {
                const float e = (float)mye[M * 1024 + kt * 16 + li];
                atomicAdd(&Sloc[j], e * rinv[v] * vk);      // ds_add_f32
            }
        }
    }
    __syncthreads();
    for (int i = tid; i < 1024; i += 256)
        atomicAdd(&Sg[(size_t)hb * L_ + i], Sloc[i]);       // global_atomic_add_f32
}

// ---------------------------------------------------------------------------
// Kernel 4: out[b,j,h] = (S[j-1]+S[j]+S[j+1]) / cnt[j]  (SAME window, W=3)
// ---------------------------------------------------------------------------
__global__ __launch_bounds__(256) void final_k(
    const float* __restrict__ S, float* __restrict__ out)
{
    const int t = blockIdx.x * 256 + threadIdx.x;    // < B*L*H, out is (B,L,H) row-major
    const int b = t >> 13;
    const int j = (t >> 3) & 1023;
    const int h = t & 7;
    const float* s = S + (size_t)((h * B_ + b) << 10);
    float acc = s[j], cnt = 1.0f;
    if (j > 0)       { acc += s[j - 1];   cnt += 1.0f; }
    if (j < L_ - 1)  { acc += s[j + 1];   cnt += 1.0f; }
    out[t] = acc / cnt;
}

// ---------------------------------------------------------------------------
// Host-side launcher
// ---------------------------------------------------------------------------
extern "C" void kernel_launch(void* const* d_in, const int* in_sizes, int n_in,
                              void* d_out, int out_size, void* d_ws, size_t ws_size,
                              hipStream_t stream)
{
    const float* x  = (const float*)d_in[0];   // (B,L,D)
    const float* Wq = (const float*)d_in[1];   // (D, H*HD)
    const float* bq = (const float*)d_in[2];   // (H,1,1,HD) -> 512 flat
    const float* Wk = (const float*)d_in[3];   // (D, H*HD)
    const float* bk = (const float*)d_in[4];   // (H,1,1,HD)
    const float* Wv = (const float*)d_in[5];   // (D, H)
    const float* pe = (const float*)d_in[6];   // (L, D)

    char* ws = (char*)d_ws;
    // Workspace layout (≈10.75 MB total):
    _Float16* Ah   = (_Float16*)(ws);                         //  2 MB   xpe f16
    _Float16* Wh   = (_Float16*)(ws + (2u << 20));            //  0.5 MB [Wq|Wk]^T f16
    _Float16* Cb   = (_Float16*)(ws + 2621440u);              //  8 MB   [Q|K] f16
    float*    vbuf = (float*)   (ws + 11010048u);             //  128 KB v'
    float*    Sg   = (float*)   (ws + 11141120u);             //  128 KB diagonal sums

    prep_kernel<<<4096, 256, 0, stream>>>(x, pe, Wq, Wk, Ah, Wh, Sg);
    gemm_qk  <<<1024, 128, 0, stream>>>(Ah, Wh, bq, bk, Cb);
    v_kernel <<<128, 256, 0, stream>>>(x, Wv, vbuf);

    const size_t lds_bytes = 8u * 16u * 1024u * 2u + 1024u * 4u;   // 260 KB (<= 320 KB/WGP)
    attn_kernel<<<dim3(H_ * B_, L_ / 128), 256, lds_bytes, stream>>>(Cb, vbuf, Sg);

    final_k<<<128, 256, 0, stream>>>(Sg, (float*)d_out);
}
